// DimixLoss_neg_10849087390440
// MI455X (gfx1250) — matrix operand order repaired and checked by reading, not compiled, not run on hardware
//
#include <hip/hip_runtime.h>
#include <hip/hip_bf16.h>

// ---------------------------------------------------------------------------
// DimixLoss fused pipeline for MI455X (gfx1250, wave32, WMMA)
//   B=16, F=2048, N=H*W=64, k=F/2=1024
//   K1: center + L2-normalize (fp32 -> f16 operands, L2-resident, 8 MB)
//   K2: per 16-row strip: WMMA YX = X'M'^T + M'X'^T in LDS, then
//       radix-select top-k statistic entirely in LDS (no S/P in HBM)
//   K3: global min + exp-mean -> scalar
// ---------------------------------------------------------------------------

typedef __attribute__((ext_vector_type(16))) _Float16 v16h;
typedef __attribute__((ext_vector_type(8)))  float    v8f;

#define B_   16
#define F_   2048
#define N_   64
#define KTOP 1024

// ------------------------- helpers -----------------------------------------

union FragU { uint4 u[2]; v16h h; };

// A-fragment (16x32 f16, M x K) per CDNA5 ISA layout:
//   lanes 0-15 : row M=lane,   VGPR0-3 = K 0..7,  VGPR4-7 = K 16..23
//   lanes16-31 : row M=lane-16,VGPR0-3 = K 8..15, VGPR4-7 = K 24..31
// With row-major [row][64 halves] storage this is two contiguous 16B chunks.
__device__ inline v16h load_a_frag(const _Float16* rowbase, int h, int lane) {
    int kb = (lane & 16) ? 8 : 0;
    int o0 = h * 32 + kb;                 // first chunk start (halves)
    const uint4* p = (const uint4*)rowbase;
    FragU f;
    f.u[0] = p[(o0 >> 3)];                // K = o0 .. o0+7
    f.u[1] = p[(o0 >> 3) + 2];            // K = o0+16 .. o0+23
    return f.h;
}

// B-fragment (32x16 f16, K x N): column N = lane&15 comes from row (g0+N) of
// the (row-major) operand; lanes 0-15 hold K=0..7,16..23, lanes 16-31 hold K=8..15,24..31.
// Two 16B chunks per lane (same K order as the A-fragment).
__device__ inline v16h load_b_frag(const _Float16* rowbase, int h, int lane) {
    int o = h * 32 + ((lane & 16) ? 8 : 0);
    const uint4* p = (const uint4*)rowbase;
    FragU f;
    f.u[0] = p[(o >> 3)];
    f.u[1] = p[(o >> 3) + 2];
    return f.h;
}

__device__ inline unsigned fkey(float f) {     // order-preserving f32 -> u32
    unsigned u = __float_as_uint(f);
    return (u & 0x80000000u) ? ~u : (u | 0x80000000u);
}
__device__ inline float keyf(unsigned k) {     // inverse
    unsigned u = (k & 0x80000000u) ? (k & 0x7fffffffu) : ~k;
    return __uint_as_float(u);
}

// ------------------- kernel 1: center + L2-normalize -> f16 -----------------

__global__ void __launch_bounds__(256)
norm_f16_kernel(const float* __restrict__ X, const float* __restrict__ M,
                _Float16* __restrict__ Xh, _Float16* __restrict__ Mh) {
    int b = blockIdx.x;
    int t = threadIdx.x;
    int n = t & 63, chunk = t >> 6;
    const float* Xb = X + (size_t)b * F_ * N_;
    const float* Mb = M + (size_t)b * F_ * N_;

    float sx = 0.f, sxx = 0.f, sm = 0.f, smm = 0.f;
    for (int f = chunk * 512; f < chunk * 512 + 512; ++f) {
        float x = Xb[f * N_ + n];
        float m = Mb[f * N_ + n];
        sx += x; sxx += x * x; sm += m; smm += m * m;
    }
    __shared__ float red[4][64][4];
    red[chunk][n][0] = sx;  red[chunk][n][1] = sxx;
    red[chunk][n][2] = sm;  red[chunk][n][3] = smm;
    __syncthreads();

    __shared__ float meanX[64], rstdX[64], meanM[64], rstdM[64];
    if (t < 64) {
        float a0 = 0.f, a1 = 0.f, a2 = 0.f, a3 = 0.f;
        for (int c = 0; c < 4; ++c) {
            a0 += red[c][t][0]; a1 += red[c][t][1];
            a2 += red[c][t][2]; a3 += red[c][t][3];
        }
        float mx = a0 / (float)F_;
        float nx = sqrtf(fmaxf(a1 - (float)F_ * mx * mx, 0.f));
        meanX[t] = mx; rstdX[t] = 1.f / (nx + 1e-10f);
        float mm = a2 / (float)F_;
        float nm = sqrtf(fmaxf(a3 - (float)F_ * mm * mm, 0.f));
        meanM[t] = mm; rstdM[t] = 1.f / (nm + 1e-10f);
    }
    __syncthreads();

    _Float16* Xhb = Xh + (size_t)b * F_ * N_;
    _Float16* Mhb = Mh + (size_t)b * F_ * N_;
    for (int i = t; i < F_ * N_; i += 256) {
        int nn = i & 63;
        Xhb[i] = (_Float16)((Xb[i] - meanX[nn]) * rstdX[nn]);
        Mhb[i] = (_Float16)((Mb[i] - meanM[nn]) * rstdM[nn]);
    }
}

// ------- kernel 2: fused WMMA GEMM + radix-select top-k statistic -----------
// grid: (F_/16, B_), 256 threads (8 waves). Each block owns 16 rows of YX.

__global__ void __launch_bounds__(256)
fused_yx_topk_kernel(const _Float16* __restrict__ Xh,
                     const _Float16* __restrict__ Mh,
                     float* __restrict__ Cxy) {
    __shared__ unsigned yxk[16][F_];    // 128 KB: YX strip, later order-keys
    __shared__ unsigned hist[16][256];  // 16 KB: per-row radix histograms
    __shared__ unsigned redu[256];
    __shared__ float    redf[256];
    __shared__ float    redf2[256];
    __shared__ unsigned scur[16];       // selected key prefix per row
    __shared__ int      sknd[16];       // remaining k within prefix per row
    __shared__ unsigned smaxk[16];      // row max (key space)

    int b  = blockIdx.y;
    int f0 = blockIdx.x * 16;
    int tid = threadIdx.x, lane = tid & 31, wave = tid >> 5;

    const _Float16* Xb = Xh + (size_t)b * F_ * N_;
    const _Float16* Mb = Mh + (size_t)b * F_ * N_;

    // ---- phase 1: WMMA GEMM, fragments straight from global (L2-hit). -----
    const _Float16* arX = Xb + (size_t)(f0 + (lane & 15)) * N_;
    const _Float16* arM = Mb + (size_t)(f0 + (lane & 15)) * N_;
    v16h aX0 = load_a_frag(arX, 0, lane), aX1 = load_a_frag(arX, 1, lane);
    v16h aM0 = load_a_frag(arM, 0, lane), aM1 = load_a_frag(arM, 1, lane);

    for (int tIdx = wave; tIdx < F_ / 16; tIdx += 8) {   // uniform trip count
        int g0 = tIdx * 16;
        const _Float16* brM = Mb + (size_t)(g0 + (lane & 15)) * N_;
        const _Float16* brX = Xb + (size_t)(g0 + (lane & 15)) * N_;
        v8f c = {0.f, 0.f, 0.f, 0.f, 0.f, 0.f, 0.f, 0.f};
        // YX[f,g] = x_f . m_g + m_f . x_g   (K=64 -> two K=32 steps each)
        c = __builtin_amdgcn_wmma_f32_16x16x32_f16(false, aX0, false,
                load_b_frag(brM, 0, lane), (short)0, c, false, false);
        c = __builtin_amdgcn_wmma_f32_16x16x32_f16(false, aX1, false,
                load_b_frag(brM, 1, lane), (short)0, c, false, false);
        c = __builtin_amdgcn_wmma_f32_16x16x32_f16(false, aM0, false,
                load_b_frag(brX, 0, lane), (short)0, c, false, false);
        c = __builtin_amdgcn_wmma_f32_16x16x32_f16(false, aM1, false,
                load_b_frag(brX, 1, lane), (short)0, c, false, false);
        // D layout: VGPR r, lanes0-15 -> M=r, lanes16-31 -> M=8+r; N=lane&15
        int rbase = (lane & 16) ? 8 : 0;
        int col   = g0 + (lane & 15);
#pragma unroll
        for (int r = 0; r < 8; ++r) yxk[rbase + r][col] = __float_as_uint(c[r]);
    }
    __syncthreads();

    // ---- convert to order-preserving keys once, in place -------------------
    {
        unsigned* flat = &yxk[0][0];
        for (int i = tid; i < 16 * F_; i += 256) {
            unsigned u = flat[i];
            flat[i] = (u & 0x80000000u) ? ~u : (u | 0x80000000u);
        }
    }
    __syncthreads();

    // ---- phase 2: per-row top-k statistic. 16 threads per row. -------------
    int row  = tid >> 4, slot = tid & 15;
    int rowg = f0 + row;                 // global f index (for dist)

    // row max in key space (softmax Z cancels; max only for exp stability)
    unsigned mk = 0;
    for (int j = slot; j < F_; j += 16) mk = mk > yxk[row][j] ? mk : yxk[row][j];
    redu[tid] = mk;
    __syncthreads();
    if (slot == 0) {
        unsigned m = redu[tid];
        for (int s = 1; s < 16; ++s) m = m > redu[tid + s] ? m : redu[tid + s];
        smaxk[row] = m;
    }
    __syncthreads();
    float mx = keyf(smaxk[row]);

    // 8-bit MSB-first radix select for the k-th largest key (4 passes)
    unsigned prefix = 0;
    int kneed = KTOP;
#pragma unroll
    for (int d = 3; d >= 0; --d) {
        int shift = d * 8;
        for (int i = slot; i < 256; i += 16) hist[row][i] = 0u;
        __syncthreads();
        for (int j = slot; j < F_; j += 16) {
            unsigned key = yxk[row][j];
            bool ok = (d == 3) || ((key >> (shift + 8)) == prefix);
            if (ok) atomicAdd(&hist[row][(key >> shift) & 255u], 1u);
        }
        __syncthreads();
        // chunk partial sums (16 bins per slot), then slot0 locates the bin
        unsigned csum = 0;
        for (int i = 0; i < 16; ++i) csum += hist[row][slot * 16 + i];
        redu[tid] = csum;
        __syncthreads();
        if (slot == 0) {
            int kk = kneed;
            unsigned cum = 0;
            int csel = 0;
            for (int c = 15; c >= 0; --c) {
                unsigned cs = redu[row * 16 + c];
                if (cum + cs >= (unsigned)kk) { csel = c; break; }
                cum += cs;
            }
            int bsel = csel * 16;
            for (int i = 15; i >= 0; --i) {
                unsigned hv = hist[row][csel * 16 + i];
                if (cum + hv >= (unsigned)kk) { bsel = csel * 16 + i; break; }
                cum += hv;
            }
            scur[row] = (prefix << 8) | (unsigned)bsel;
            sknd[row] = kk - (int)cum;   // still needed inside selected bin
        }
        __syncthreads();
        prefix = scur[row];
        kneed  = sknd[row];
    }
    unsigned cur = prefix;               // exact key of the k-th largest
    int r = kneed;                       // #ties at cur to include (>=1)
    float tval = keyf(cur);

    // accumulate over the strictly-greater set
    float se = 0.f, sed = 0.f;
    for (int j = slot; j < F_; j += 16) {
        unsigned kk = yxk[row][j];
        if (kk > cur) {
            float e = __expf(keyf(kk) - mx);
            se  += e;
            sed += e * fabsf((float)(j - rowg));
        }
    }
    redf[tid] = se; redf2[tid] = sed;
    __syncthreads();
    if (slot == 0) {
        float s1 = 0.f, s2 = 0.f;
        for (int s = 0; s < 16; ++s) { s1 += redf[tid + s]; s2 += redf2[tid + s]; }
        float te = __expf(tval - mx);
        // ties included lowest-index-first (matches jax.lax.top_k tie-break)
        int taken = 0; float tds = 0.f;
        for (int j = 0; j < F_ && taken < r; ++j) {
            if (yxk[row][j] == cur) { tds += fabsf((float)(j - rowg)); ++taken; }
        }
        float sumV  = s1 + (float)r * te;
        float sumVD = s2 + te * tds;
        Cxy[b * F_ + rowg] = (sumVD / sumV) / (float)KTOP;
    }
}

// --------------------- kernel 3: global min + exp-mean ----------------------

__global__ void __launch_bounds__(256)
finalize_kernel(const float* __restrict__ Cxy, float* __restrict__ out) {
    __shared__ float red[256];
    int t = threadIdx.x;

    float mn = 3.4e38f;
    for (int i = t; i < B_ * F_; i += 256) mn = fminf(mn, Cxy[i]);
    red[t] = mn;
    __syncthreads();
    for (int s = 128; s > 0; s >>= 1) {
        if (t < s) red[t] = fminf(red[t], red[t + s]);
        __syncthreads();
    }
    float gmin = red[0];
    __syncthreads();

    float sum = 0.f;
    for (int i = t; i < B_ * F_; i += 256) sum += expf(-Cxy[i] + gmin - 1e-6f);
    red[t] = sum;
    __syncthreads();
    for (int s = 128; s > 0; s >>= 1) {
        if (t < s) red[t] += red[t + s];
        __syncthreads();
    }
    if (t == 0) out[0] = red[0] / (float)(B_ * F_);
}

// ----------------------------- launcher -------------------------------------

extern "C" void kernel_launch(void* const* d_in, const int* in_sizes, int n_in,
                              void* d_out, int out_size, void* d_ws, size_t ws_size,
                              hipStream_t stream) {
    const float* X = (const float*)d_in[0];
    const float* M = (const float*)d_in[1];
    float* out = (float*)d_out;

    char* ws = (char*)d_ws;
    const size_t opBytes = (size_t)B_ * F_ * N_ * sizeof(_Float16); // 4 MB
    _Float16* Xh  = (_Float16*)(ws);
    _Float16* Mh  = (_Float16*)(ws + opBytes);
    float*    Cxy = (float*)   (ws + 2 * opBytes);                  // 128 KB

    norm_f16_kernel<<<B_, 256, 0, stream>>>(X, M, Xh, Mh);

    dim3 grid(F_ / 16, B_);
    fused_yx_topk_kernel<<<grid, 256, 0, stream>>>(Xh, Mh, Cxy);

    finalize_kernel<<<1, 256, 0, stream>>>(Cxy, out);
}
